// SelfAttention_28956669510293
// MI455X (gfx1250) — matrix operand order repaired and checked by reading, not compiled, not run on hardware
//
#include <hip/hip_runtime.h>
#include <hip/hip_bf16.h>

// B=4, H=W=64, C=256 -> 4096 tokens/batch, 16384 total. CQK=32, CV=128.
// Flash-attention fused pipeline, all matmuls on v_wmma_f32_16x16x32_bf16.
// Attention K/V blocks staged in LDS via async global->LDS copies (double buffered).

typedef __attribute__((ext_vector_type(16))) __bf16 v16bf;
typedef __attribute__((ext_vector_type(8)))  float  v8f;
typedef int v4i __attribute__((vector_size(16)));   // matches async-LDS builtin param type

union Frag {
    uint4 u[2];   // two 16B chunks per lane = 32B = 16 bf16
    v16bf v;
};

__device__ __forceinline__ unsigned short f2bf(float f) {
    unsigned int u = __float_as_uint(f);
    u = (u + 0x7FFFu + ((u >> 16) & 1u)) >> 16;   // round-to-nearest-even
    return (unsigned short)u;
}

__device__ __forceinline__ v8f wmma_bf16(const v16bf& a, const v16bf& b, const v8f& c) {
    return __builtin_amdgcn_wmma_f32_16x16x32_bf16(false, a, false, b, (short)0, c, false, false);
}

// 16-byte global -> LDS copy: async (ASYNCcnt-tracked) when the gfx1250 builtin
// exists, else synchronous load+ds_store (loop structure stays correct).
__device__ __forceinline__ void copy16_g2l(const unsigned short* g, unsigned short* l) {
#if __has_builtin(__builtin_amdgcn_global_load_async_to_lds_b128)
    __builtin_amdgcn_global_load_async_to_lds_b128(
        (__attribute__((address_space(1))) v4i*)g,
        (__attribute__((address_space(3))) v4i*)l, 0, 0);
#else
    *(uint4*)l = *(const uint4*)g;
#endif
}

template <int N>
__device__ __forceinline__ void wait_async() {
#if __has_builtin(__builtin_amdgcn_s_wait_asynccnt)
    __builtin_amdgcn_s_wait_asynccnt(N);
#else
#if __has_builtin(__builtin_amdgcn_global_load_async_to_lds_b128)
    asm volatile("s_wait_asynccnt %0" ::"n"(N) : "memory");
#endif
#endif
}

// ---------------------------------------------------------------- converts
__global__ __launch_bounds__(256) void convert_x_kernel(const float* __restrict__ x,
                                                        unsigned short* __restrict__ Xb, int n4) {
    int i = blockIdx.x * 256 + threadIdx.x;
    if (i < n4) {
        float4 v = ((const float4*)x)[i];
        ushort4 o;
        o.x = f2bf(v.x); o.y = f2bf(v.y); o.z = f2bf(v.z); o.w = f2bf(v.w);
        ((ushort4*)Xb)[i] = o;
    }
}

// Transposed bf16 weights: WfT/WgT [32][256], WhT [128][256], WoT [256][128]
__global__ __launch_bounds__(256) void convert_w_kernel(
    const float* __restrict__ Wf, const float* __restrict__ Wg,
    const float* __restrict__ Wh, const float* __restrict__ Wo,
    unsigned short* __restrict__ WfT, unsigned short* __restrict__ WgT,
    unsigned short* __restrict__ WhT, unsigned short* __restrict__ WoT) {
    int i = blockIdx.x * 256 + threadIdx.x;                 // 81920 total
    if (i < 8192) {                                         // WfT[n][k] = Wf[k][n]
        int n = i >> 8, k = i & 255;
        WfT[i] = f2bf(Wf[k * 32 + n]);
    } else if (i < 16384) {
        int ii = i - 8192; int n = ii >> 8, k = ii & 255;
        WgT[ii] = f2bf(Wg[k * 32 + n]);
    } else if (i < 49152) {                                 // WhT[128][256]
        int ii = i - 16384; int n = ii >> 8, k = ii & 255;
        WhT[ii] = f2bf(Wh[k * 128 + n]);
    } else if (i < 81920) {                                 // WoT[256][128]
        int ii = i - 49152; int n = ii >> 7, k = ii & 127;
        WoT[ii] = f2bf(Wo[k * 256 + n]);
    }
}

// ---------------------------------------------------------------- projections
// Per wave: 16 tokens. F,G row-major [16384][32]; V channel-major Vt[b][128][4096].
__global__ __launch_bounds__(256) void proj_kernel(
    const unsigned short* __restrict__ Xb,
    const unsigned short* __restrict__ WfT, const unsigned short* __restrict__ WgT,
    const unsigned short* __restrict__ WhT,
    const float* __restrict__ bf_, const float* __restrict__ bg_, const float* __restrict__ bh_,
    unsigned short* __restrict__ Fb, unsigned short* __restrict__ Gb,
    unsigned short* __restrict__ Vt) {
    const int wave = threadIdx.x >> 5, lane = threadIdx.x & 31;
    const int hi = lane >> 4, ln = lane & 15;
    const int t0 = (blockIdx.x * 8 + wave) * 16;            // token-tile base

    v8f accF[2], accG[2], accV[8];
    const v8f z8 = {0.f,0.f,0.f,0.f,0.f,0.f,0.f,0.f};
#pragma unroll
    for (int j = 0; j < 2; ++j) { accF[j] = z8; accG[j] = z8; }
#pragma unroll
    for (int j = 0; j < 8; ++j) accV[j] = z8;

    for (int kc = 0; kc < 256; kc += 32) {
        Frag aX;                                            // A: 16x32 bf16
        const unsigned short* px = Xb + (size_t)(t0 + ln) * 256 + kc + hi * 16;
        aX.u[0] = *(const uint4*)(px);
        aX.u[1] = *(const uint4*)(px + 8);
#pragma unroll
        for (int j = 0; j < 2; ++j) {
            Frag bw;
            const unsigned short* p = WfT + (size_t)(j * 16 + ln) * 256 + kc + hi * 16;
            bw.u[0] = *(const uint4*)(p); bw.u[1] = *(const uint4*)(p + 8);
            accF[j] = wmma_bf16(aX.v, bw.v, accF[j]);
        }
#pragma unroll
        for (int j = 0; j < 2; ++j) {
            Frag bw;
            const unsigned short* p = WgT + (size_t)(j * 16 + ln) * 256 + kc + hi * 16;
            bw.u[0] = *(const uint4*)(p); bw.u[1] = *(const uint4*)(p + 8);
            accG[j] = wmma_bf16(aX.v, bw.v, accG[j]);
        }
#pragma unroll
        for (int j = 0; j < 8; ++j) {
            Frag bw;
            const unsigned short* p = WhT + (size_t)(j * 16 + ln) * 256 + kc + hi * 16;
            bw.u[0] = *(const uint4*)(p); bw.u[1] = *(const uint4*)(p + 8);
            accV[j] = wmma_bf16(aX.v, bw.v, accV[j]);
        }
    }
#pragma unroll
    for (int j = 0; j < 2; ++j)
#pragma unroll
        for (int r = 0; r < 8; ++r) {
            int tok = t0 + r + 8 * hi, col = j * 16 + ln;
            Fb[(size_t)tok * 32 + col] = f2bf(accF[j][r] + bf_[col]);
            Gb[(size_t)tok * 32 + col] = f2bf(accG[j][r] + bg_[col]);
        }
#pragma unroll
    for (int j = 0; j < 8; ++j)
#pragma unroll
        for (int r = 0; r < 8; ++r) {
            int tok = t0 + r + 8 * hi, col = j * 16 + ln;
            int bb = tok >> 12, ti = tok & 4095;
            Vt[((size_t)bb * 128 + col) * 4096 + ti] = f2bf(accV[j][r] + bh_[col]);
        }
}

// ---------------------------------------------------------------- flash attention
// Block = 8 waves = 128 queries; key blocks of 64 staged in LDS (double buffered,
// async global->LDS). Per wave: 16 queries, 8x(16x16) f32 accumulators over d_v=128.
__global__ __launch_bounds__(256) void attn_kernel(
    const unsigned short* __restrict__ Gb, const unsigned short* __restrict__ Fb,
    const unsigned short* __restrict__ Vt, unsigned short* __restrict__ Ob) {
    // padded rows: F 32->40, V 64->72, P 64->72 (bank-conflict-free b128 reads)
    __shared__ __align__(16) unsigned short Fs[2][64][40];    // 10 KB
    __shared__ __align__(16) unsigned short Vs[2][128][72];   // 36 KB
    __shared__ __align__(16) unsigned short Ps[8][16][72];    // 18 KB

    const int tid = threadIdx.x;
    const int wave = tid >> 5, lane = tid & 31;
    const int hi = lane >> 4, ln = lane & 15;
    const int wg = blockIdx.x;                   // 128 blocks
    const int b = wg >> 5;                       // batch 0..3
    const int q0 = ((wg & 31) * 8 + wave) * 16;  // query base within batch

    const unsigned short* Gq = Gb + (size_t)(b * 4096 + q0) * 32;
    const unsigned short* Fk = Fb + (size_t)b * 4096 * 32;
    const unsigned short* Vb = Vt + (size_t)b * 128 * 4096;

    // cooperative stage copy: F block 64x32 (1 op/thread), V block 128x64 (4 ops/thread)
    auto issue_stage = [&](int kb, int buf) {
        {
            int row = tid >> 2, c = tid & 3;
            copy16_g2l(Fk + (size_t)(kb + row) * 32 + c * 8, &Fs[buf][row][c * 8]);
        }
        {
            int row = tid >> 1, h = tid & 1;
            const unsigned short* g = Vb + (size_t)row * 4096 + kb + h * 32;
            unsigned short* l = &Vs[buf][row][h * 32];
#pragma unroll
            for (int c = 0; c < 4; ++c) copy16_g2l(g + c * 8, l + c * 8);
        }
    };

    Frag aG;                                     // persistent query A-fragment
    {
        const unsigned short* p = Gq + ln * 32 + hi * 16;
        aG.u[0] = *(const uint4*)(p);
        aG.u[1] = *(const uint4*)(p + 8);
    }

    const v8f z8 = {0.f,0.f,0.f,0.f,0.f,0.f,0.f,0.f};
    v8f accO[8];
    float m[8], l_[8];
#pragma unroll
    for (int t = 0; t < 8; ++t) accO[t] = z8;
#pragma unroll
    for (int r = 0; r < 8; ++r) { m[r] = -__builtin_inff(); l_[r] = 0.f; }

    issue_stage(0, 0);                           // prologue: stages 0 and 1 in flight
    issue_stage(64, 1);

    const int NIT = 4096 / 64;
    for (int it = 0; it < NIT; ++it) {
        const int buf = it & 1;
        if (it + 1 < NIT) wait_async<5>();       // stage `it` done, `it+1` in flight
        else              wait_async<0>();
        __syncthreads();                         // all waves' stage data visible

        // ---- S = G . F^T : four 16-key tiles, K=32 -> one WMMA each
        v8f sfr[4];
#pragma unroll
        for (int s = 0; s < 4; ++s) {
            Frag bF;
            const unsigned short* p = &Fs[buf][s * 16 + ln][hi * 16];
            bF.u[0] = *(const uint4*)(p); bF.u[1] = *(const uint4*)(p + 8);
            sfr[s] = wmma_bf16(aG.v, bF.v, z8);
        }

        // ---- online softmax (row stats via half-wave xor reductions)
        float sc[8];
#pragma unroll
        for (int r = 0; r < 8; ++r) {
            float cmax = fmaxf(fmaxf(sfr[0][r], sfr[1][r]), fmaxf(sfr[2][r], sfr[3][r]));
#pragma unroll
            for (int off = 1; off < 16; off <<= 1)
                cmax = fmaxf(cmax, __shfl_xor(cmax, off, 32));
            float mnew = fmaxf(m[r], cmax);
            float rs = 0.f;
#pragma unroll
            for (int s = 0; s < 4; ++s) {
                float pe = __expf(sfr[s][r] - mnew);
                rs += pe;
                Ps[wave][r + 8 * hi][s * 16 + ln] = f2bf(pe);  // P tile, D-layout -> LDS
            }
#pragma unroll
            for (int off = 1; off < 16; off <<= 1)
                rs += __shfl_xor(rs, off, 32);
            sc[r] = __expf(m[r] - mnew);
            l_[r] = l_[r] * sc[r] + rs;
            m[r] = mnew;
        }
#pragma unroll
        for (int t = 0; t < 8; ++t)
#pragma unroll
            for (int r = 0; r < 8; ++r)
                accO[t][r] *= sc[r];

        // same-wave cross-lane LDS handoff of P: DS counter wait, no barrier needed
        asm volatile("s_wait_dscnt 0" ::: "memory");

        Frag aP0, aP1;                           // P as two 16x32 A-fragments
        {
            const unsigned short* p = &Ps[wave][ln][hi * 16];
            aP0.u[0] = *(const uint4*)(p);      aP0.u[1] = *(const uint4*)(p + 8);
            aP1.u[0] = *(const uint4*)(p + 32); aP1.u[1] = *(const uint4*)(p + 40);
        }

        // ---- O += P . V  (8 channel tiles x 2 K-chunks from LDS V stage)
#pragma unroll
        for (int t = 0; t < 8; ++t) {
            Frag bV0, bV1;
            const unsigned short* p = &Vs[buf][t * 16 + ln][hi * 16];
            bV0.u[0] = *(const uint4*)(p);      bV0.u[1] = *(const uint4*)(p + 8);
            bV1.u[0] = *(const uint4*)(p + 32); bV1.u[1] = *(const uint4*)(p + 40);
            accO[t] = wmma_bf16(aP0.v, bV0.v, accO[t]);
            accO[t] = wmma_bf16(aP1.v, bV1.v, accO[t]);
        }

        __syncthreads();                         // all waves done reading `buf`
        if (it + 2 < NIT) issue_stage((it + 2) * 64, buf);
    }

    // ---- epilogue: normalize, store bf16 O [16384][128]
    unsigned short* Orow = Ob + (size_t)(b * 4096 + q0) * 128;
#pragma unroll
    for (int r = 0; r < 8; ++r) {
        float inv = 1.f / l_[r];
#pragma unroll
        for (int t = 0; t < 8; ++t)
            Orow[(size_t)(r + 8 * hi) * 128 + t * 16 + ln] = f2bf(accO[t][r] * inv);
    }
}

// ---------------------------------------------------------------- output projection + residual
// Wave = 16 tokens x 128 out-channels (half). out = x + O @ Wo + bo.
__global__ __launch_bounds__(256) void outproj_kernel(
    const unsigned short* __restrict__ Ob, const unsigned short* __restrict__ WoT,
    const float* __restrict__ bo, const float* __restrict__ x, float* __restrict__ out) {
    const int wave = threadIdx.x >> 5, lane = threadIdx.x & 31;
    const int hi = lane >> 4, ln = lane & 15;
    const int t0 = (blockIdx.x * 4 + (wave >> 1)) * 16;     // token tile
    const int nh = wave & 1;                                // channel half

    const v8f z8 = {0.f,0.f,0.f,0.f,0.f,0.f,0.f,0.f};
    v8f acc[8];
#pragma unroll
    for (int j = 0; j < 8; ++j) acc[j] = z8;

    for (int kc = 0; kc < 128; kc += 32) {
        Frag aO;
        const unsigned short* p = Ob + (size_t)(t0 + ln) * 128 + kc + hi * 16;
        aO.u[0] = *(const uint4*)(p);
        aO.u[1] = *(const uint4*)(p + 8);
#pragma unroll
        for (int j = 0; j < 8; ++j) {
            Frag bw;
            const unsigned short* pw = WoT + (size_t)(nh * 128 + j * 16 + ln) * 128 + kc + hi * 16;
            bw.u[0] = *(const uint4*)(pw); bw.u[1] = *(const uint4*)(pw + 8);
            acc[j] = wmma_bf16(aO.v, bw.v, acc[j]);
        }
    }
#pragma unroll
    for (int j = 0; j < 8; ++j)
#pragma unroll
        for (int r = 0; r < 8; ++r) {
            int tok = t0 + r + 8 * hi, col = nh * 128 + j * 16 + ln;
            size_t idx = (size_t)tok * 256 + col;
            out[idx] = x[idx] + acc[j][r] + bo[col];
        }
}

// ---------------------------------------------------------------- launch
extern "C" void kernel_launch(void* const* d_in, const int* in_sizes, int n_in,
                              void* d_out, int out_size, void* d_ws, size_t ws_size,
                              hipStream_t stream) {
    const float* x  = (const float*)d_in[0];
    const float* Wf = (const float*)d_in[1];
    const float* bf = (const float*)d_in[2];
    const float* Wg = (const float*)d_in[3];
    const float* bg = (const float*)d_in[4];
    const float* Wh = (const float*)d_in[5];
    const float* bh = (const float*)d_in[6];
    const float* Wo = (const float*)d_in[7];
    const float* bo = (const float*)d_in[8];
    float* out = (float*)d_out;

    char* ws = (char*)d_ws;
    size_t off = 0;
    auto take = [&](size_t bytes) {
        char* p = ws + off;
        off += (bytes + 255) & ~(size_t)255;
        return p;
    };
    unsigned short* Xb  = (unsigned short*)take((size_t)16384 * 256 * 2);  // 8 MB
    unsigned short* WfT = (unsigned short*)take((size_t)32 * 256 * 2);
    unsigned short* WgT = (unsigned short*)take((size_t)32 * 256 * 2);
    unsigned short* WhT = (unsigned short*)take((size_t)128 * 256 * 2);
    unsigned short* WoT = (unsigned short*)take((size_t)256 * 128 * 2);
    unsigned short* Fb  = (unsigned short*)take((size_t)16384 * 32 * 2);   // 1 MB
    unsigned short* Gb  = (unsigned short*)take((size_t)16384 * 32 * 2);   // 1 MB
    unsigned short* Vt  = (unsigned short*)take((size_t)4 * 128 * 4096 * 2); // 4 MB
    unsigned short* Ob  = (unsigned short*)take((size_t)16384 * 128 * 2);  // 4 MB

    convert_x_kernel<<<4096, 256, 0, stream>>>(x, Xb, 1048576);
    convert_w_kernel<<<320, 256, 0, stream>>>(Wf, Wg, Wh, Wo, WfT, WgT, WhT, WoT);
    proj_kernel<<<128, 256, 0, stream>>>(Xb, WfT, WgT, WhT, bf, bg, bh, Fb, Gb, Vt);
    attn_kernel<<<128, 256, 0, stream>>>(Gb, Fb, Vt, Ob);
    outproj_kernel<<<256, 256, 0, stream>>>(Ob, WoT, bo, x, out);
}
